// CLSFewShotClassifier_17111149707608
// MI455X (gfx1250) — matrix operand order repaired and checked by reading, not compiled, not run on hardware
//
#include <hip/hip_runtime.h>

// ---------------------------------------------------------------------------
// Few-shot matching-network classifier for MI455X (gfx1250).
// Conv4 backbone as implicit-GEMM on v_wmma_f32_16x16x32_f16 (wave32 WMMA),
// weights staged into LDS by the Tensor Data Mover (tensor_load_to_lds,
// double-buffered, s_wait_tensorcnt), f16 activations with f32 accumulation.
// Prototype + cosine head on VALU (memory-bound tail).
// ---------------------------------------------------------------------------

typedef __attribute__((ext_vector_type(16))) _Float16 v16h;
typedef __attribute__((ext_vector_type(8)))  float    v8f;

#define NB    8     // tasks
#define NS    25    // support per task
#define NT    75    // targets per task
#define NC    5     // classes
#define NIMG  800   // NB*(NS+NT)
#define NSUP  200   // NB*NS
#define EDIM  2304  // 64*6*6
#define FEPS  1e-8f

#if __has_builtin(__builtin_amdgcn_tensor_load_to_lds)
#define HAVE_TDM 1
typedef __attribute__((ext_vector_type(4))) unsigned int u32x4;
typedef __attribute__((ext_vector_type(8))) int          i32x8;
typedef __attribute__((ext_vector_type(4))) int          i32x4;

// TDM: load a [rows x 64] f16 tile (row-major, stride 64) from global into LDS.
// D# layout per CDNA5 ISA (2D tensor, groups 2/3 zero).
// This toolchain exposes the 6-arg builtin:
//   (uint32x4 g0, int32x8 g1, int32x4, int32x4, int32x8, i32 cpol)
__device__ __forceinline__ void tdm_load_b_chunk(const _Float16* gsrc,
                                                 void* ldsDst, int rows) {
    unsigned long long ga = (unsigned long long)(size_t)gsrc;
    unsigned int       la = (unsigned int)(size_t)ldsDst;  // LDS byte offset
    u32x4 g0;
    g0[0] = 1u;                                             // count=1, user mode
    g0[1] = la;                                             // lds_addr
    g0[2] = (unsigned int)ga;                               // global_addr[31:0]
    g0[3] = (unsigned int)((ga >> 32) & 0x1FFFFFFull)       // global_addr[56:32]
          | (2u << 30);                                     // type = 2 (image)
    i32x8 g1;
    g1[0] = (1 << 16);                // data_size = 1 -> 2 bytes; mask/flags 0
    g1[1] = (64 << 16);               // tensor_dim0 = 64   (bits 79:48)
    g1[2] = (rows & 0xFFFF) << 16;    // tensor_dim1 lo16   (bits 111:80)
    g1[3] = (64 << 16);               // tensor_dim1 hi = 0 | tile_dim0 = 64
    g1[4] = rows;                     // tile_dim1 = rows, tile_dim2 = 0
    g1[5] = 64;                       // tensor_dim0_stride = 64 (bits 207:160)
    g1[6] = 0;
    g1[7] = 0;                        // tensor_dim1_stride = 0 (2D)
    i32x4 gz4 = {0, 0, 0, 0};
    i32x8 gz8 = {0, 0, 0, 0, 0, 0, 0, 0};
    __builtin_amdgcn_tensor_load_to_lds(g0, g1, gz4, gz4, gz8, 0);
}
#endif

// ---------------------------------------------------------------------------
// Weight packing: W [64][Cin][3][3] (f32, OIHW) -> Bmat [Kpad][64] (f16),
// k = cin*9 + kh*3 + kw, zero-padded rows for k >= Cin*9.
// ---------------------------------------------------------------------------
__global__ void pack_weights_kernel(const float* __restrict__ W,
                                    _Float16* __restrict__ Bmat,
                                    int Cin, int Kpad) {
    int idx = blockIdx.x * blockDim.x + threadIdx.x;
    int total = Kpad * 64;
    if (idx >= total) return;
    int k = idx >> 6;
    int n = idx & 63;
    float v = 0.0f;
    if (k < Cin * 9) {
        int cin = k / 9;
        int r   = k % 9;
        v = W[((n * Cin + cin) * 3 + (r / 3)) * 3 + (r % 3)];
    }
    Bmat[idx] = (_Float16)v;
}

// ---------------------------------------------------------------------------
// Implicit-GEMM conv3x3 stride-2 (+bias, +ReLU) via WMMA.
//   M = HOUT*WOUT output pixels, N = 64 output channels, K = CIN*9.
//   Block: 128 threads = 4 waves; wave w owns N columns [16w, 16w+16) and two
//   16-row M subtiles (block M-tile = 32). B chunks (32x64 f16) are staged
//   into double-buffered LDS by the TDM one chunk ahead of the WMMAs.
//   Grid: (ceil(M/32), NIMG).
// Fragment layouts follow CDNA5 ISA §7.12.2.
// ---------------------------------------------------------------------------
template <typename TIn, int CIN, int HIN, int WIN, int HOUT, int WOUT,
          int PAD, bool FINAL>
__global__ void conv_wmma_kernel(const TIn* __restrict__ inA,   // images [0, split)
                                 const TIn* __restrict__ inB,   // images [split, NIMG)
                                 int split,
                                 const _Float16* __restrict__ Bmat, // [Kpad][64]
                                 const float* __restrict__ bias,    // [64]
                                 _Float16* __restrict__ outAct,     // [NIMG][64][M] f16
                                 float* __restrict__ outEmb) {      // [NIMG][2304] f32
    constexpr int K    = CIN * 9;
    constexpr int KCH  = (K + 31) / 32;    // 32-wide K chunks
    constexpr int M    = HOUT * WOUT;
    constexpr int IPIX = HIN * WIN;

    const int img   = blockIdx.y;
    const int m0    = blockIdx.x * 32;
    const int tid   = threadIdx.x;
    const int lane  = tid & 31;
    const int wave  = tid >> 5;
    const int ncol0 = wave * 16;

    const TIn* in = (img < split)
                        ? (inA + (size_t)img * CIN * IPIX)
                        : (inB + (size_t)(img - split) * CIN * IPIX);

    __shared__ _Float16 Bsh[2][32 * 64];   // double-buffered weight chunks (8 KB)
    __shared__ _Float16 Atile[32 * 32];    // 32 pixels x 32 K-slices (2 KB)

    v8f acc[2] = {v8f{}, v8f{}};

#ifdef HAVE_TDM
    if (wave == 0)                         // prefetch chunk 0 via TDM
        tdm_load_b_chunk(Bmat, &Bsh[0][0], 32);
#endif

    for (int kc = 0; kc < KCH; ++kc) {
        const int buf = kc & 1;
        __syncthreads();   // protect Atile/Bsh reuse from previous iteration

        // Cooperative im2col of the 32x32 A tile (1024 f16, 8 per thread).
#pragma unroll
        for (int i = 0; i < 8; ++i) {
            int e  = tid + i * 128;
            int ml = e >> 5;          // local output-pixel row 0..31
            int kk = e & 31;          // local K 0..31
            int m  = m0 + ml;
            int k  = kc * 32 + kk;
            float val = 0.0f;
            if (m < M && k < K) {
                int cin = k / 9;
                int r   = k % 9;
                int oy  = m / WOUT;
                int ox  = m % WOUT;
                int iy  = oy * 2 + (r / 3) - PAD;
                int ix  = ox * 2 + (r % 3) - PAD;
                if (iy >= 0 && iy < HIN && ix >= 0 && ix < WIN)
                    val = (float)in[(size_t)cin * IPIX + iy * WIN + ix];
            }
            Atile[ml * 32 + kk] = (_Float16)val;
        }

#ifdef HAVE_TDM
        if (wave == 0) {
            if (kc + 1 < KCH) {        // issue next chunk, overlap with WMMAs
                tdm_load_b_chunk(Bmat + (size_t)(kc + 1) * 32 * 64,
                                 &Bsh[buf ^ 1][0], 32);
                __builtin_amdgcn_s_wait_tensorcnt(1);  // chunk kc complete
            } else {
                __builtin_amdgcn_s_wait_tensorcnt(0);
            }
        }
#else
        // Fallback: cooperative global->LDS copy of this weight chunk.
        for (int i = tid; i < 32 * 64; i += 128)
            Bsh[buf][i] = Bmat[(size_t)kc * 32 * 64 + i];
        if (kc + 1 < KCH)
            __builtin_prefetch(Bmat + (size_t)(kc + 1) * 32 * 64 + tid * 16, 0, 1);
#endif
        __syncthreads();   // A tile + B chunk visible to all waves

        // A fragments: 16-bit A 16x32 layout (ISA 7.12.2), two M subtiles.
        v16h a0, a1;
        {
            const int mrow = lane & 15;
            const int half = lane >> 4;
#pragma unroll
            for (int vv = 0; vv < 8; ++vv) {
                int kb = ((vv < 4) ? 0 : 16) + (vv & 3) * 2 + half * 8;
                a0[2 * vv]     = Atile[mrow * 32 + kb];
                a0[2 * vv + 1] = Atile[mrow * 32 + kb + 1];
                a1[2 * vv]     = Atile[(16 + mrow) * 32 + kb];
                a1[2 * vv + 1] = Atile[(16 + mrow) * 32 + kb + 1];
            }
        }
        // B fragment: lane holds column n = lane%16, 16 K values in the same order as A.
        v16h bfrag;
#pragma unroll
        for (int s = 0; s < 16; ++s) {
            int kb = ((s < 8) ? 0 : 16) + (s & 7) + (lane >> 4) * 8;
            bfrag[s] = Bsh[buf][kb * 64 + ncol0 + (lane & 15)];
        }

        acc[0] = __builtin_amdgcn_wmma_f32_16x16x32_f16(
            false, a0, false, bfrag, (short)0, acc[0], false, false);
        acc[1] = __builtin_amdgcn_wmma_f32_16x16x32_f16(
            false, a1, false, bfrag, (short)0, acc[1], false, false);
    }

    // Epilogue: 32-bit C/D 16x16 layout — VGPR v: lanes0-15 M=v, lanes16-31 M=v+8.
    const int n     = (lane & 15) + ncol0;
    const int mbase = (lane >> 4) * 8;
    const float bs  = bias[n];
#pragma unroll
    for (int sub = 0; sub < 2; ++sub) {
#pragma unroll
        for (int vv = 0; vv < 8; ++vv) {
            int m = m0 + sub * 16 + vv + mbase;
            if (m < M) {
                float val = fmaxf(acc[sub][vv] + bs, 0.0f);
                if constexpr (FINAL) {
                    // flatten [64][6][6] -> d = n*36 + m
                    outEmb[(size_t)img * EDIM + n * (HOUT * WOUT) + m] = val;
                } else {
                    outAct[((size_t)img * 64 + n) * M + m] = (_Float16)val;
                }
            }
        }
    }
}

// ---------------------------------------------------------------------------
// Prototypes: one block per (task b, class c).
//   proto[b][c][d] = (1/CAP) * sum_{s: y[b][s]%C==c} emb[b*S+s][d]
// Also emits ||proto|| per (b,c).
// ---------------------------------------------------------------------------
__global__ void protos_kernel(const float* __restrict__ emb,   // [NIMG][EDIM], support first
                              const int* __restrict__ y,       // [NB][NS]
                              float* __restrict__ protos,      // [NB*NC][EDIM]
                              float* __restrict__ pnorm) {     // [NB*NC]
    const int b = blockIdx.x / NC;
    const int c = blockIdx.x % NC;
    const int tid = threadIdx.x;

    __shared__ unsigned char memb[NS];
    if (tid < NS) memb[tid] = (unsigned char)((y[b * NS + tid] % NC) == c);
    __syncthreads();

    float nrm = 0.0f;
    for (int d = tid; d < EDIM; d += blockDim.x) {
        float s = 0.0f;
#pragma unroll
        for (int si = 0; si < NS; ++si)
            if (memb[si]) s += emb[(size_t)(b * NS + si) * EDIM + d];
        s *= (1.0f / (NS / NC));   // /CAP
        protos[(size_t)blockIdx.x * EDIM + d] = s;
        nrm += s * s;
    }

    __shared__ float red[256];
    red[tid] = nrm;
    __syncthreads();
    for (int s = 128; s > 0; s >>= 1) {
        if (tid < s) red[tid] += red[tid + s];
        __syncthreads();
    }
    if (tid == 0) pnorm[blockIdx.x] = sqrtf(red[0]);
}

// ---------------------------------------------------------------------------
// Cosine logits: one block per (task b, target t).
//   preds[b][t][c] = <e, p_c> / (max(||e||,eps) * max(||p_c||,eps))
// ---------------------------------------------------------------------------
__global__ void logits_kernel(const float* __restrict__ emb,    // [NIMG][EDIM]
                              const float* __restrict__ protos, // [NB*NC][EDIM]
                              const float* __restrict__ pnorm,  // [NB*NC]
                              float* __restrict__ out) {        // [NB][NT][NC]
    const int b = blockIdx.x / NT;
    const int t = blockIdx.x % NT;
    const int tid = threadIdx.x;

    const float* e = emb + (size_t)(NSUP + b * NT + t) * EDIM;

    float dot[NC] = {0.f, 0.f, 0.f, 0.f, 0.f};
    float nn = 0.0f;
    for (int d = tid; d < EDIM; d += blockDim.x) {
        float ev = e[d];
        nn += ev * ev;
#pragma unroll
        for (int c = 0; c < NC; ++c)
            dot[c] += ev * protos[(size_t)(b * NC + c) * EDIM + d];
    }

    __shared__ float red[(NC + 1) * 256];
#pragma unroll
    for (int c = 0; c < NC; ++c) red[c * 256 + tid] = dot[c];
    red[NC * 256 + tid] = nn;
    __syncthreads();
    for (int s = 128; s > 0; s >>= 1) {
        if (tid < s) {
#pragma unroll
            for (int j = 0; j < NC + 1; ++j)
                red[j * 256 + tid] += red[j * 256 + tid + s];
        }
        __syncthreads();
    }
    if (tid < NC) {
        float tn = fmaxf(sqrtf(red[NC * 256]), FEPS);
        float pn = fmaxf(pnorm[b * NC + tid], FEPS);
        out[((size_t)blockIdx.x) * NC + tid] = red[tid * 256] / (tn * pn);
    }
}

// ---------------------------------------------------------------------------
// Host-side launcher.
// ---------------------------------------------------------------------------
extern "C" void kernel_launch(void* const* d_in, const int* in_sizes, int n_in,
                              void* d_out, int out_size, void* d_ws, size_t ws_size,
                              hipStream_t stream) {
    (void)in_sizes; (void)n_in; (void)out_size; (void)ws_size;

    const float* xs = (const float*)d_in[0];   // [8,25,3,84,84]
    const float* xt = (const float*)d_in[1];   // [8,75,3,84,84]
    const int*   y  = (const int*)d_in[2];     // [8,25]
    const float* W1 = (const float*)d_in[3];
    const float* b1 = (const float*)d_in[4];
    const float* W2 = (const float*)d_in[5];
    const float* b2 = (const float*)d_in[6];
    const float* W3 = (const float*)d_in[7];
    const float* b3 = (const float*)d_in[8];
    const float* W4 = (const float*)d_in[9];
    const float* b4 = (const float*)d_in[10];

    // Carve workspace (aligned 256B).
    char*  base = (char*)d_ws;
    size_t off  = 0;
    auto take = [&](size_t bytes) -> void* {
        void* p = base + off;
        off += (bytes + 255) & ~(size_t)255;
        return p;
    };
    _Float16* act1  = (_Float16*)take((size_t)NIMG * 64 * 42 * 42 * 2); // 180.6 MB
    _Float16* act2  = (_Float16*)take((size_t)NIMG * 64 * 21 * 21 * 2); //  45.2 MB
    _Float16* act3  = (_Float16*)take((size_t)NIMG * 64 * 11 * 11 * 2); //  12.4 MB
    float*    emb   = (float*)take((size_t)NIMG * EDIM * 4);            //   7.4 MB
    float*    protos= (float*)take((size_t)NB * NC * EDIM * 4);
    float*    pnorm = (float*)take((size_t)NB * NC * 4);
    _Float16* Wb1   = (_Float16*)take((size_t)32 * 64 * 2);
    _Float16* Wb2   = (_Float16*)take((size_t)576 * 64 * 2);
    _Float16* Wb3   = (_Float16*)take((size_t)576 * 64 * 2);
    _Float16* Wb4   = (_Float16*)take((size_t)576 * 64 * 2);

    // Pack weights to f16 B-matrices.
    pack_weights_kernel<<<(32 * 64 + 255) / 256, 256, 0, stream>>>(W1, Wb1, 3, 32);
    pack_weights_kernel<<<(576 * 64 + 255) / 256, 256, 0, stream>>>(W2, Wb2, 64, 576);
    pack_weights_kernel<<<(576 * 64 + 255) / 256, 256, 0, stream>>>(W3, Wb3, 64, 576);
    pack_weights_kernel<<<(576 * 64 + 255) / 256, 256, 0, stream>>>(W4, Wb4, 64, 576);

    // Conv4 backbone (implicit GEMM on WMMA, TDM-staged weights).
    // L1: 3x84x84 -> 64x42x42  (M=1764, K=27->32, pad_lo=0)
    conv_wmma_kernel<float, 3, 84, 84, 42, 42, 0, false>
        <<<dim3((1764 + 31) / 32, NIMG), 128, 0, stream>>>(
            xs, xt, NSUP, Wb1, b1, act1, nullptr);
    // L2: 64x42x42 -> 64x21x21 (M=441, K=576, pad_lo=0)
    conv_wmma_kernel<_Float16, 64, 42, 42, 21, 21, 0, false>
        <<<dim3((441 + 31) / 32, NIMG), 128, 0, stream>>>(
            act1, act1, 0, Wb2, b2, act2, nullptr);
    // L3: 64x21x21 -> 64x11x11 (M=121, K=576, pad_lo=1)
    conv_wmma_kernel<_Float16, 64, 21, 21, 11, 11, 1, false>
        <<<dim3((121 + 31) / 32, NIMG), 128, 0, stream>>>(
            act2, act2, 0, Wb3, b3, act3, nullptr);
    // L4: 64x11x11 -> 64x6x6   (M=36, K=576, pad_lo=1) -> f32 embeddings
    conv_wmma_kernel<_Float16, 64, 11, 11, 6, 6, 1, true>
        <<<dim3((36 + 31) / 32, NIMG), 128, 0, stream>>>(
            act3, act3, 0, Wb4, b4, nullptr, emb);

    // Prototypes + norms, then cosine logits.
    protos_kernel<<<NB * NC, 256, 0, stream>>>(emb, y, protos, pnorm);
    logits_kernel<<<NB * NT, 256, 0, stream>>>(emb, protos, pnorm, (float*)d_out);
}
